// SolubilityModel_74199855005862
// MI455X (gfx1250) — compile-verified, matrix operand order checked by reading, not run on hardware
//
#include <hip/hip_runtime.h>
#include <hip/hip_bf16.h>
#include <math.h>

// ---------------------------------------------------------------------------
// Types for CDNA5 WMMA
// ---------------------------------------------------------------------------
typedef __attribute__((ext_vector_type(16))) _Float16 v16h;
typedef __attribute__((ext_vector_type(8)))  float    v8f;

#define HC    64
#define HEADS 4
#define HID   16
#define EDIM  16
#define BN_EPS 1e-5f
#define XS    68   // padded LDS row stride (floats): bank = (4m+k)%64, conflict-free

// ---------------------------------------------------------------------------
// CDNA5 async global->LDS copy (VGLOBAL GLOBAL_LOAD_ASYNC_TO_LDS_B128,
// tracked by ASYNCcnt). LDS operand is the byte offset within the wave's
// LDS allocation (single __shared__ array -> offset base 0).
// ---------------------------------------------------------------------------
__device__ __forceinline__ void async_copy_b128(unsigned lds_byte_off,
                                                const float* gptr) {
  unsigned long long ga = (unsigned long long)(const void*)gptr;
  asm volatile("global_load_async_to_lds_b128 %0, %1, off"
               :: "v"(lds_byte_off), "v"(ga)
               : "memory");
}
__device__ __forceinline__ void wait_async0() {
  asm volatile("s_wait_asynccnt 0x0" ::: "memory");
}

// ---------------------------------------------------------------------------
// WMMA fragment loaders (wave32, V_WMMA_F32_16X16X32_F16 layouts per ISA 7.12.2)
// A (16x32 f16, MxK): lanes 0-15 -> row M=lane, K pattern {kb..kb+7, kb+16..23},
// kb=0; lanes 16-31 -> row M=lane-16, kb=8; (k,k+1) pairs packed per VGPR.
// B (32x16, KxN) mirrors A with each lane holding one column.
// ---------------------------------------------------------------------------
__device__ __forceinline__ v16h fragA_lds(const float* __restrict__ sx, int k0) {
  const int lane = threadIdx.x & 31;
  const int m    = lane & 15;
  const int kb   = (lane < 16) ? 0 : 8;
  const float* rp = sx + m * XS;
  v16h f;
#pragma unroll
  for (int half = 0; half < 2; ++half) {
#pragma unroll
    for (int i = 0; i < 4; ++i) {
      const int k = k0 + 16 * half + kb + 2 * i;
      f[half * 8 + 2 * i]     = (_Float16)rp[k];
      f[half * 8 + 2 * i + 1] = (_Float16)rp[k + 1];
    }
  }
  return f;
}

__device__ __forceinline__ v16h load_fragB_f32(const float* __restrict__ W, int ldw,
                                               int k0, int c0) {
  const int lane = threadIdx.x & 31;
  const int n    = lane & 15;
  const int kb   = (lane < 16) ? 0 : 8;
  v16h f;
#pragma unroll
  for (int half = 0; half < 2; ++half) {
#pragma unroll
    for (int i = 0; i < 4; ++i) {
      const int k = k0 + 16 * half + kb + 2 * i;
      f[half * 8 + 2 * i]     = (_Float16)W[(size_t)k * ldw + c0 + n];
      f[half * 8 + 2 * i + 1] = (_Float16)W[(size_t)(k + 1) * ldw + c0 + n];
    }
  }
  return f;
}

// ---------------------------------------------------------------------------
// K1: node-side GEMMs. One block = 4 waves; wave w computes the 16x16 tile of
// output matrix w (Q,K,V,Skip); the 16x64 f32 x-tile is staged to LDS once via
// async b128 copies and shared by all 4 waves. Grid = (ceil(N/16), 4 col-tiles).
// ---------------------------------------------------------------------------
__global__ void __launch_bounds__(128)
gemm_node_qkvs(const float* __restrict__ xin, int Nn,
               const float* __restrict__ W0, const float* __restrict__ W1,
               const float* __restrict__ W2, const float* __restrict__ W3,
               const float* __restrict__ b0, const float* __restrict__ b1,
               const float* __restrict__ b2, const float* __restrict__ b3,
               float* __restrict__ o0, float* __restrict__ o1,
               float* __restrict__ o2, float* __restrict__ o3) {
  __shared__ float sx[16 * XS];

  const int tid  = threadIdx.x;
  const int row0 = blockIdx.x * 16;
  const int c0   = blockIdx.y * 16;
  const bool full = (row0 + 16 <= Nn);

  if (full) {
    // 16 rows x 16 float4-chunks = 256 async b128 copies, 2 per thread.
#pragma unroll
    for (int it = 0; it < 2; ++it) {
      const int ch   = tid + it * 128;       // 0..255
      const int r    = ch >> 4;
      const int cseg = ch & 15;
      const unsigned lds = (unsigned)((r * XS + cseg * 4) * 4);
      async_copy_b128(lds, xin + (size_t)(row0 + r) * HC + cseg * 4);
    }
    wait_async0();
  } else {
    for (int i = tid; i < 16 * HC; i += 128) {
      const int r = i >> 6, cc = i & 63;
      sx[r * XS + cc] = (row0 + r < Nn) ? xin[(size_t)(row0 + r) * HC + cc] : 0.0f;
    }
  }
  __syncthreads();

  const int wave = tid >> 5;   // 0..3 -> Q,K,V,S
  const float* W = (wave == 0) ? W0 : (wave == 1) ? W1 : (wave == 2) ? W2 : W3;
  const float* b = (wave == 0) ? b0 : (wave == 1) ? b1 : (wave == 2) ? b2 : b3;
  float*       o = (wave == 0) ? o0 : (wave == 1) ? o1 : (wave == 2) ? o2 : o3;

  const v16h a0  = fragA_lds(sx, 0);
  const v16h a1  = fragA_lds(sx, 32);
  const v16h bb0 = load_fragB_f32(W, HC, 0,  c0);
  const v16h bb1 = load_fragB_f32(W, HC, 32, c0);

  v8f c = {};
  c = __builtin_amdgcn_wmma_f32_16x16x32_f16(false, a0, false, bb0, (short)0, c, false, false);
  c = __builtin_amdgcn_wmma_f32_16x16x32_f16(false, a1, false, bb1, (short)0, c, false, false);

  const int lane = tid & 31;
  const int n    = lane & 15;
  const int mb   = (lane < 16) ? 0 : 8;
  const float bias = b[c0 + n];
  float* op = o + (size_t)(row0 + mb) * HC + c0 + n;
  if (full) {
#pragma unroll
    for (int r = 0; r < 8; ++r) op[(size_t)r * HC] = c[r] + bias;
  } else {
#pragma unroll
    for (int r = 0; r < 8; ++r)
      if (row0 + mb + r < Nn) op[(size_t)r * HC] = c[r] + bias;
  }
}

// ---------------------------------------------------------------------------
// Per-layer init of attention accumulators.
// ---------------------------------------------------------------------------
__global__ void init_layer_accum(float* __restrict__ agg, float* __restrict__ nmax,
                                 float* __restrict__ nsum, int Nn) {
  const int i = blockIdx.x * blockDim.x + threadIdx.x;
  if (i < Nn * HC) agg[i] = 0.0f;
  if (i < Nn * HEADS) { nmax[i] = -INFINITY; nsum[i] = 0.0f; }
}

__device__ __forceinline__ void atomicMaxFloat(float* addr, float v) {
  unsigned int* a = (unsigned int*)addr;
  unsigned int old = __hip_atomic_load(a, __ATOMIC_RELAXED, __HIP_MEMORY_SCOPE_AGENT);
  while (__uint_as_float(old) < v) {
    unsigned int assumed = old;
    old = atomicCAS(a, assumed, __float_as_uint(v));
    if (old == assumed) break;
  }
}

// ---------------------------------------------------------------------------
// K2: logits[e,h] = dot(q[dst,h,:], k[src,h,:] + (edge_attr[e] @ We)[h,:]) / 4
// e_emb recomputed on the fly from the 4KB LDS-resident We tile (cheaper than
// streaming a 205MB materialized e_emb through L2/HBM twice).
// ---------------------------------------------------------------------------
__global__ void __launch_bounds__(256)
edge_logits(const int* __restrict__ ei, const float* __restrict__ ea,
            const float* __restrict__ q, const float* __restrict__ k,
            const float* __restrict__ We, float* __restrict__ logits,
            float* __restrict__ nmax, int Ee) {
  __shared__ float sWe[EDIM * HC];
  for (int i = threadIdx.x; i < EDIM * HC; i += blockDim.x) sWe[i] = We[i];
  __syncthreads();

  const int t = blockIdx.x * blockDim.x + threadIdx.x;
  if (t >= Ee * HEADS) return;
  const int e = t >> 2, h = t & 3;
  const int s = ei[e], d = ei[Ee + e];

  const float* qp = q  + (size_t)d * HC + h * HID;
  const float* kp = k  + (size_t)s * HC + h * HID;
  const float* ap = ea + (size_t)e * EDIM;

  float av[EDIM];
#pragma unroll
  for (int j = 0; j < EDIM; ++j) av[j] = ap[j];

  float acc = 0.0f;
#pragma unroll
  for (int c = 0; c < HID; ++c) {
    float ee = 0.0f;
#pragma unroll
    for (int j = 0; j < EDIM; ++j) ee += av[j] * sWe[j * HC + h * HID + c];
    acc += qp[c] * (kp[c] + ee);
  }
  const float lg = acc * 0.25f;                 // 1/sqrt(HID)
  logits[t] = lg;
  atomicMaxFloat(&nmax[d * HEADS + h], lg);
}

// K3: ex = exp(logit - max[dst]); accumulate denominator.
__global__ void __launch_bounds__(256)
edge_expsum(const int* __restrict__ ei, float* __restrict__ logits,
            const float* __restrict__ nmax, float* __restrict__ nsum, int Ee) {
  const int t = blockIdx.x * blockDim.x + threadIdx.x;
  if (t >= Ee * HEADS) return;
  const int e = t >> 2, h = t & 3;
  const int d = ei[Ee + e];
  const float ex = __expf(logits[t] - nmax[d * HEADS + h]);
  logits[t] = ex;
  atomicAdd(&nsum[d * HEADS + h], ex);
}

// K4: agg[dst] += alpha * (v[src] + e_emb), alpha = ex / sum[dst].
__global__ void __launch_bounds__(256)
edge_aggregate(const int* __restrict__ ei, const float* __restrict__ ea,
               const float* __restrict__ v, const float* __restrict__ We,
               const float* __restrict__ logits, const float* __restrict__ nsum,
               float* __restrict__ agg, int Ee) {
  __shared__ float sWe[EDIM * HC];
  for (int i = threadIdx.x; i < EDIM * HC; i += blockDim.x) sWe[i] = We[i];
  __syncthreads();

  const int t = blockIdx.x * blockDim.x + threadIdx.x;
  if (t >= Ee * HEADS) return;
  const int e = t >> 2, h = t & 3;
  const int s = ei[e], d = ei[Ee + e];

  const float alpha = logits[t] / nsum[d * HEADS + h];
  const float* vp = v  + (size_t)s * HC + h * HID;
  const float* ap = ea + (size_t)e * EDIM;
  float* op = agg + (size_t)d * HC + h * HID;

  float av[EDIM];
#pragma unroll
  for (int j = 0; j < EDIM; ++j) av[j] = ap[j];

#pragma unroll
  for (int c = 0; c < HID; ++c) {
    float ee = 0.0f;
#pragma unroll
    for (int j = 0; j < EDIM; ++j) ee += av[j] * sWe[j * HC + h * HID + c];
    atomicAdd(&op[c], alpha * (vp[c] + ee));
  }
}

// K5: x = relu(BN(agg + skip))
__global__ void __launch_bounds__(256)
post_bn_relu(const float* __restrict__ agg, const float* __restrict__ skip,
             const float* __restrict__ g, const float* __restrict__ b,
             const float* __restrict__ m, const float* __restrict__ vv,
             float* __restrict__ xout, int Nn) {
  const int i = blockIdx.x * blockDim.x + threadIdx.x;
  if (i >= Nn * HC) return;
  const int c = i & (HC - 1);
  const float val = agg[i] + skip[i];
  const float y = (val - m[c]) * rsqrtf(vv[c] + BN_EPS) * g[c] + b[c];
  xout[i] = fmaxf(y, 0.0f);
}

// Pooling: per-graph mean via L2 atomics.
__global__ void pool_zero(float* __restrict__ pooled, float* __restrict__ counts, int Gg) {
  const int i = blockIdx.x * blockDim.x + threadIdx.x;
  if (i < Gg * HC) pooled[i] = 0.0f;
  if (i < Gg) counts[i] = 0.0f;
}

__global__ void __launch_bounds__(256)
pool_sum(const float* __restrict__ x, const int* __restrict__ batch,
         float* __restrict__ pooled, float* __restrict__ counts, int Nn) {
  const int i = blockIdx.x * blockDim.x + threadIdx.x;
  if (i >= Nn * HC) return;
  const int n = i >> 6, c = i & (HC - 1);
  const int g = batch[n];
  atomicAdd(&pooled[(size_t)g * HC + c], x[i]);
  if (c == 0) atomicAdd(&counts[g], 1.0f);
}

// ---------------------------------------------------------------------------
// Head FC1 via WMMA: hmid = relu((pooled/count) @ fc1_w + b1).
// [G,64]@[64,16]; mean-division folded into the A-fragment f32->f16 convert.
// One wave per 16-row tile; G=2000 -> 125 full tiles.
// ---------------------------------------------------------------------------
__global__ void __launch_bounds__(32)
head_fc1_wmma(const float* __restrict__ pooled, const float* __restrict__ counts,
              const float* __restrict__ w, const float* __restrict__ b,
              float* __restrict__ hmid, int Gg) {
  const int row0 = blockIdx.x * 16;
  const int lane = threadIdx.x & 31;
  const int m    = lane & 15;
  const int kb   = (lane < 16) ? 0 : 8;
  const int row  = row0 + m;
  const bool rv  = (row < Gg);
  const float inv = rv ? 1.0f / fmaxf(counts[row], 1.0f) : 0.0f;
  const float* rp = pooled + (size_t)row * HC;

  v16h a0, a1;
#pragma unroll
  for (int half = 0; half < 2; ++half) {
#pragma unroll
    for (int i = 0; i < 4; ++i) {
      const int k = 16 * half + kb + 2 * i;
      a0[half * 8 + 2 * i]     = (_Float16)(rv ? rp[k]      * inv : 0.0f);
      a0[half * 8 + 2 * i + 1] = (_Float16)(rv ? rp[k + 1]  * inv : 0.0f);
      a1[half * 8 + 2 * i]     = (_Float16)(rv ? rp[32 + k]     * inv : 0.0f);
      a1[half * 8 + 2 * i + 1] = (_Float16)(rv ? rp[32 + k + 1] * inv : 0.0f);
    }
  }
  const v16h b0 = load_fragB_f32(w, HID, 0,  0);
  const v16h b1 = load_fragB_f32(w, HID, 32, 0);

  v8f c = {};
  c = __builtin_amdgcn_wmma_f32_16x16x32_f16(false, a0, false, b0, (short)0, c, false, false);
  c = __builtin_amdgcn_wmma_f32_16x16x32_f16(false, a1, false, b1, (short)0, c, false, false);

  const int n  = lane & 15;
  const int mb = (lane < 16) ? 0 : 8;
  const float bias = b[n];
#pragma unroll
  for (int r = 0; r < 8; ++r) {
    const int orow = row0 + mb + r;
    if (orow < Gg) hmid[(size_t)orow * HID + n] = fmaxf(c[r] + bias, 0.0f);
  }
}

__global__ void __launch_bounds__(256)
head_fc2(const float* __restrict__ hmid, const float* __restrict__ w,
         const float* __restrict__ b, float* __restrict__ out, int Gg) {
  const int g = blockIdx.x * blockDim.x + threadIdx.x;
  if (g >= Gg) return;
  float acc = b[0];
#pragma unroll
  for (int j = 0; j < HID; ++j) acc += hmid[(size_t)g * HID + j] * w[j];
  out[g] = acc;
}

// ---------------------------------------------------------------------------
// Host-side orchestration
// ---------------------------------------------------------------------------
extern "C" void kernel_launch(void* const* d_in, const int* in_sizes, int n_in,
                              void* d_out, int out_size, void* d_ws, size_t ws_size,
                              hipStream_t stream) {
  const float* x     = (const float*)d_in[0];
  const int*   ei    = (const int*)  d_in[1];
  const float* ea    = (const float*)d_in[2];
  const int*   batch = (const int*)  d_in[3];
  const float* Wq = (const float*)d_in[4];
  const float* bq = (const float*)d_in[5];
  const float* Wk = (const float*)d_in[6];
  const float* bk = (const float*)d_in[7];
  const float* Wv = (const float*)d_in[8];
  const float* bv = (const float*)d_in[9];
  const float* We = (const float*)d_in[10];
  const float* Ws = (const float*)d_in[11];
  const float* bs = (const float*)d_in[12];
  const float* bn_g = (const float*)d_in[13];
  const float* bn_b = (const float*)d_in[14];
  const float* bn_m = (const float*)d_in[15];
  const float* bn_v = (const float*)d_in[16];
  const float* fc1_w = (const float*)d_in[17];
  const float* fc1_b = (const float*)d_in[18];
  const float* fc2_w = (const float*)d_in[19];
  const float* fc2_b = (const float*)d_in[20];

  const int N = in_sizes[0] / HC;
  const int E = in_sizes[1] / 2;
  const int L = in_sizes[4] / (HC * HC);
  const int G = out_size;

  // Scratch layout (floats)
  float* ws = (float*)d_ws;
  size_t o = 0;
  float* bufx  = ws + o; o += (size_t)N * HC;
  float* qb    = ws + o; o += (size_t)N * HC;
  float* kb    = ws + o; o += (size_t)N * HC;
  float* vb    = ws + o; o += (size_t)N * HC;
  float* sk    = ws + o; o += (size_t)N * HC;
  float* agg   = ws + o; o += (size_t)N * HC;
  float* lgts  = ws + o; o += (size_t)E * HEADS;
  float* nmax  = ws + o; o += (size_t)N * HEADS;
  float* nsum  = ws + o; o += (size_t)N * HEADS;
  float* pooled= ws + o; o += (size_t)G * HC;
  float* counts= ws + o; o += (size_t)G;
  float* hmid  = ws + o; o += (size_t)G * HID;

  const int nrt = (N + 15) / 16;
  const int eThreads = E * HEADS;
  const int eBlocks  = (eThreads + 255) / 256;
  const int nBlocks  = (N * HC + 255) / 256;

  for (int l = 0; l < L; ++l) {
    const float* xin = (l == 0) ? x : bufx;
    const size_t wOff = (size_t)l * HC * HC;
    const size_t bOff = (size_t)l * HC;

    gemm_node_qkvs<<<dim3(nrt, HC / 16), 128, 0, stream>>>(
        xin, N,
        Wq + wOff, Wk + wOff, Wv + wOff, Ws + wOff,
        bq + bOff, bk + bOff, bv + bOff, bs + bOff,
        qb, kb, vb, sk);

    init_layer_accum<<<nBlocks, 256, 0, stream>>>(agg, nmax, nsum, N);

    edge_logits<<<eBlocks, 256, 0, stream>>>(
        ei, ea, qb, kb, We + (size_t)l * EDIM * HC, lgts, nmax, E);

    edge_expsum<<<eBlocks, 256, 0, stream>>>(ei, lgts, nmax, nsum, E);

    edge_aggregate<<<eBlocks, 256, 0, stream>>>(
        ei, ea, vb, We + (size_t)l * EDIM * HC, lgts, nsum, agg, E);

    post_bn_relu<<<nBlocks, 256, 0, stream>>>(
        agg, sk, bn_g + bOff, bn_b + bOff, bn_m + bOff, bn_v + bOff, bufx, N);
  }

  pool_zero<<<(G * HC + 255) / 256, 256, 0, stream>>>(pooled, counts, G);
  pool_sum<<<nBlocks, 256, 0, stream>>>(bufx, batch, pooled, counts, N);
  head_fc1_wmma<<<(G + 15) / 16, 32, 0, stream>>>(pooled, counts, fc1_w, fc1_b, hmid, G);
  head_fc2<<<(G + 255) / 256, 256, 0, stream>>>(hmid, fc2_w, fc2_b, (float*)d_out, G);
}
